// ConGNN_6597069767264
// MI455X (gfx1250) — matrix operand order repaired and checked by reading, not compiled, not run on hardware
//
#include <hip/hip_runtime.h>
#include <math.h>

// ---------------- problem constants (from reference) ----------------
#define NF_   16384   // face nodes
#define NO_   32768   // object nodes
#define BB_   128     // batch
#define DD_   512     // hidden dim
#define HH_   4       // heads
#define DH_   128     // head dim
#define KK_   3       // in-edges per node (plus self loop)
#define PFC   128     // faces per batch
#define POC   256     // objects per batch
#define MAXN_ 384     // PF + PO
#define NC_   3       // classes

// ---------------- CDNA5 WMMA types ----------------
typedef __attribute__((ext_vector_type(16))) __bf16          v16bf;
typedef __attribute__((ext_vector_type(8)))  float           v8f;
typedef __attribute__((ext_vector_type(8)))  unsigned short  us8v;

union Frag16 { v16bf bf; us8v h[2]; };

__device__ __forceinline__ unsigned short f2bf(float f) {
  unsigned u = __float_as_uint(f);
  u += 0x7FFFu + ((u >> 16) & 1u);      // round-to-nearest-even
  return (unsigned short)(u >> 16);
}

// Tensor Data Mover path (6-arg builtin on this toolchain); fallback to vector loads.
#if defined(__HIP_DEVICE_COMPILE__) && __has_builtin(__builtin_amdgcn_tensor_load_to_lds)
#define USE_TDM 1
#else
#define USE_TDM 0
#endif

#if USE_TDM
typedef __attribute__((ext_vector_type(4))) unsigned int u32x4;
typedef __attribute__((ext_vector_type(8))) int          i32x8;
typedef __attribute__((ext_vector_type(4))) int          i32x4;

// DMA a rows x cols bf16 tile (row stride = stride elems) from global into LDS.
// D# packing per CDNA5 ISA (2D tensor, groups 2/3 unused).
__device__ __forceinline__ void tdm_load_tile(const unsigned short* gptr, unsigned lds_off,
                                              int stride, int rows, int cols)
{
  unsigned long long ga = (unsigned long long)(uintptr_t)gptr;
  u32x4 g0;
  g0[0] = 1u;                                              // count=1, user descriptor
  g0[1] = lds_off;                                         // lds_addr (bytes)
  g0[2] = (unsigned)(ga & 0xFFFFFFFFull);                  // global_addr[31:0]
  g0[3] = (unsigned)((ga >> 32) & 0x1FFFFFFull) | (2u << 30); // addr[56:32] | type=2
  unsigned td0 = (unsigned)stride;                         // tensor_dim0 >= cols
  unsigned td1 = (unsigned)rows;                           // tensor_dim1
  i32x8 g1;
  g1[0] = (int)(1u << 16);                                 // data_size=1 -> 2 bytes
  g1[1] = (int)((td0 & 0xFFFFu) << 16);                    // tensor_dim0[15:0]
  g1[2] = (int)((td0 >> 16) | ((td1 & 0xFFFFu) << 16));    // td0[31:16] | td1[15:0]
  g1[3] = (int)((td1 >> 16) | ((unsigned)cols << 16));     // td1[31:16] | tile_dim0
  g1[4] = (int)(unsigned)rows;                             // tile_dim1 (tile_dim2=0)
  g1[5] = (int)(unsigned)stride;                           // tensor_dim0_stride[31:0]
  g1[6] = 0;                                               // stride hi | dim1_stride lo
  g1[7] = 0;
  i32x4 z4 = {0, 0, 0, 0};
  i32x8 z8 = {0, 0, 0, 0, 0, 0, 0, 0};
  __builtin_amdgcn_tensor_load_to_lds(g0, g1, z4, z4, z8, 0);
}
#endif

// =====================================================================
// GEMM:  C[M,N] = act( A[M,K](bf16) * Bw[N,K](bf16)^T + bias[N] )
// Requires M%128==0, N%128==0, K%32==0 (true for all shapes in this model).
// block tile 128x128, 8 waves, wave tile 32x64 -> 8 WMMAs per K-step.
// Double-buffered LDS; tiles DMA'd by the TDM (wave 0) when available.
// =====================================================================
#define TBM 128
#define TBN 128
#define TBK 32

__global__ __launch_bounds__(256) void gemm_bf16_kernel(
    const unsigned short* __restrict__ A, const unsigned short* __restrict__ Bw,
    const float* __restrict__ bias, float* __restrict__ C,
    int M, int N, int K, int act)
{
  __shared__ __align__(16) unsigned short As[2][TBM][TBK];
  __shared__ __align__(16) unsigned short Bs[2][TBN][TBK];
  const int tid  = threadIdx.x;
  const int wave = tid >> 5, lane = tid & 31;
  const int bm = blockIdx.y * TBM, bn = blockIdx.x * TBN;
  const int wm = (wave & 3) * 32, wn = (wave >> 2) * 64;
  const int lrow = lane & 15, lhi = lane >> 4;
  v8f acc[2][4] = {};

  auto load_tiles = [&](int k0, int buf) {
#if USE_TDM
    if (wave == 0) {
      tdm_load_tile(A  + (size_t)bm * K + k0, (unsigned)(uintptr_t)&As[buf][0][0], K, TBM, TBK);
      tdm_load_tile(Bw + (size_t)bn * K + k0, (unsigned)(uintptr_t)&Bs[buf][0][0], K, TBN, TBK);
    }
#else
    const int r = tid >> 1, c = (tid & 1) * 16;   // 128x32 tile, 16 elem/thread
    {
      const unsigned short* s = A + (size_t)(bm + r) * K + k0 + c;
      *(us8v*)&As[buf][r][c]     = *(const us8v*)s;
      *(us8v*)&As[buf][r][c + 8] = *(const us8v*)(s + 8);
      if (k0 + TBK < K) __builtin_prefetch(s + TBK, 0, 1);
    }
    {
      const unsigned short* s = Bw + (size_t)(bn + r) * K + k0 + c;
      *(us8v*)&Bs[buf][r][c]     = *(const us8v*)s;
      *(us8v*)&Bs[buf][r][c + 8] = *(const us8v*)(s + 8);
      if (k0 + TBK < K) __builtin_prefetch(s + TBK, 0, 1);
    }
#endif
  };

  load_tiles(0, 0);
#if USE_TDM
  if (wave == 0) __builtin_amdgcn_s_wait_tensorcnt(0);
#endif
  __syncthreads();

  const int nk = K / TBK;
  for (int t = 0; t < nk; ++t) {
    const int buf = t & 1;
    if (t + 1 < nk) load_tiles((t + 1) * TBK, buf ^ 1);

    // A fragment: lane holds row M=lane%16; K = {8h..8h+7, 16+8h..+7}, h=lane/16
    Frag16 fa[2], fb[4];
    const int ah = lhi * 8;
#pragma unroll
    for (int i = 0; i < 2; ++i) {
      fa[i].h[0] = *(const us8v*)&As[buf][wm + i * 16 + lrow][ah];
      fa[i].h[1] = *(const us8v*)&As[buf][wm + i * 16 + lrow][16 + ah];
    }
    // B fragment: lane holds col N=lane%16; K = {8h..8h+7, 16+8h..+7}, h=lane/16
    const int bk = lhi * 8;
#pragma unroll
    for (int j = 0; j < 4; ++j) {
      fb[j].h[0] = *(const us8v*)&Bs[buf][wn + j * 16 + lrow][bk];
      fb[j].h[1] = *(const us8v*)&Bs[buf][wn + j * 16 + lrow][16 + bk];
    }
#pragma unroll
    for (int i = 0; i < 2; ++i)
#pragma unroll
      for (int j = 0; j < 4; ++j)
        acc[i][j] = __builtin_amdgcn_wmma_f32_16x16x32_bf16(
            false, fa[i].bf, false, fb[j].bf, (short)0, acc[i][j], false, false);

#if USE_TDM
    if (wave == 0) __builtin_amdgcn_s_wait_tensorcnt(0);
#endif
    __syncthreads();
  }

  // epilogue: C/D layout: VGPR r -> M = r + 8*(lane>=16), N = lane%16
#pragma unroll
  for (int i = 0; i < 2; ++i)
#pragma unroll
    for (int j = 0; j < 4; ++j) {
      const int col = bn + wn + j * 16 + lrow;
      const float bv = bias ? bias[col] : 0.f;
      const int rbase = bm + wm + i * 16 + lhi * 8;
      v8f a = acc[i][j];
#pragma unroll
      for (int r = 0; r < 8; ++r) {
        float v = a[r] + bv;
        if (act == 1) v = fmaxf(v, 0.f);
        else if (act == 2) v = tanhf(v);
        C[(size_t)(rbase + r) * N + col] = v;
      }
    }
}

// =====================================================================
// LayerNorm: y = res + act( LN(x + xadd) * g + b ); optional f32/bf16 out
// act: 0 none, 1 relu, 2 elu
// =====================================================================
__global__ __launch_bounds__(256) void ln_kernel(
    const float* __restrict__ x, const float* __restrict__ xadd,
    const float* __restrict__ g, const float* __restrict__ bt,
    const float* __restrict__ res,
    float* __restrict__ yf, unsigned short* __restrict__ ybf,
    int cols, int act)
{
  const int row = blockIdx.x, tid = threadIdx.x;
  const size_t base = (size_t)row * cols;
  __shared__ float red[256];
  float s = 0.f, ss = 0.f;
  for (int c = tid; c < cols; c += 256) {
    float v = x[base + c];
    if (xadd) v += xadd[base + c];
    s += v; ss += v * v;
  }
  red[tid] = s; __syncthreads();
  for (int o = 128; o; o >>= 1) { if (tid < o) red[tid] += red[tid + o]; __syncthreads(); }
  const float mean = red[0] / (float)cols; __syncthreads();
  red[tid] = ss; __syncthreads();
  for (int o = 128; o; o >>= 1) { if (tid < o) red[tid] += red[tid + o]; __syncthreads(); }
  const float var = red[0] / (float)cols - mean * mean;
  const float inv = rsqrtf(var + 1e-5f);
  for (int c = tid; c < cols; c += 256) {
    float v = x[base + c];
    if (xadd) v += xadd[base + c];
    v = (v - mean) * inv * g[c] + bt[c];
    if (act == 1) v = fmaxf(v, 0.f);
    else if (act == 2) v = (v > 0.f) ? v : (__expf(v) - 1.f);
    if (res) v += res[base + c];
    if (yf)  yf[base + c] = v;
    if (ybf) ybf[base + c] = f2bf(v);
  }
}

// f32 -> bf16
__global__ void cvt_bf16_kernel(const float* __restrict__ x, unsigned short* __restrict__ y, size_t n) {
  size_t i = (size_t)blockIdx.x * 256 + threadIdx.x;
  if (i < n) y[i] = f2bf(x[i]);
}

// =====================================================================
// GATv2 edge logits: wave per (edge, head); lane strides head dim
// =====================================================================
__global__ __launch_bounds__(256) void gat_logit_kernel(
    const float* __restrict__ xl, const float* __restrict__ xr,
    const int* __restrict__ src, const float* __restrict__ att,
    float* __restrict__ logit, int Nn, int Ereal)
{
  int wid  = (int)(((size_t)blockIdx.x * blockDim.x + threadIdx.x) >> 5);
  int lane = threadIdx.x & 31;
  int Etot = Ereal + Nn;
  int e = wid / HH_, hh = wid - e * HH_;
  if (e >= Etot) return;
  int s, t;
  if (e < Ereal) { s = src[e]; t = e / KK_; } else { s = e - Ereal; t = s; }
  const float* pl = xl + (size_t)s * DD_ + hh * DH_;
  const float* pr = xr + (size_t)t * DD_ + hh * DH_;
  const float* pa = att + hh * DH_;
  float acc = 0.f;
  for (int d = lane; d < DH_; d += 32) {
    float v = pl[d] + pr[d];
    v = (v > 0.f) ? v : 0.2f * v;          // leaky_relu(0.2)
    acc += v * pa[d];
  }
  for (int o = 16; o; o >>= 1) acc += __shfl_xor(acc, o, 32);
  if (lane == 0) logit[(size_t)e * HH_ + hh] = acc;
}

// per-node softmax over the 4 incoming edges + weighted sum of xl[src]
__global__ __launch_bounds__(128) void gat_agg_kernel(
    const float* __restrict__ xl, const int* __restrict__ src,
    const float* __restrict__ logit, float* __restrict__ out,
    int Nn, int Ereal)
{
  const int n = blockIdx.x, tid = threadIdx.x;
  __shared__ int   sidx[4];
  __shared__ float w[HH_][4];
  if (tid < 4) sidx[tid] = (tid < KK_) ? src[n * KK_ + tid] : n;
  if (tid < 16) {
    int j = tid & 3, hh = tid >> 2;
    int e = (j < KK_) ? (n * KK_ + j) : (Ereal + n);
    w[hh][j] = logit[(size_t)e * HH_ + hh];
  }
  __syncthreads();
  if (tid < HH_) {
    float m = w[tid][0];
    for (int j = 1; j < 4; ++j) m = fmaxf(m, w[tid][j]);
    float den = 0.f;
    for (int j = 0; j < 4; ++j) { float e = __expf(w[tid][j] - m); w[tid][j] = e; den += e; }
    float inv = 1.f / (den + 1e-16f);
    for (int j = 0; j < 4; ++j) w[tid][j] *= inv;
  }
  __syncthreads();
  for (int hh = 0; hh < HH_; ++hh) {
    int col = hh * DH_ + tid;
    float acc = 0.f;
    for (int j = 0; j < 4; ++j) acc += w[hh][j] * xl[(size_t)sidx[j] * DD_ + col];
    out[(size_t)n * DD_ + col] = acc;
  }
}

// s[i] = dot(x[i,:cols], w) + b2   (wave per row)
__global__ void rowdot_kernel(const float* __restrict__ x, const float* __restrict__ w,
                              const float* __restrict__ b2, float* __restrict__ s,
                              int rows, int cols)
{
  int wid  = (int)(((size_t)blockIdx.x * blockDim.x + threadIdx.x) >> 5);
  int lane = threadIdx.x & 31;
  if (wid >= rows) return;
  float acc = 0.f;
  for (int c = lane; c < cols; c += 32) acc += x[(size_t)wid * cols + c] * w[c];
  for (int o = 16; o; o >>= 1) acc += __shfl_xor(acc, o, 32);
  if (lane == 0) s[wid] = acc + b2[0];
}

__global__ __launch_bounds__(256) void max_reduce_kernel(const float* __restrict__ s, int n, float* __restrict__ out) {
  __shared__ float red[256];
  int tid = threadIdx.x; float m = -3.4e38f;
  for (int i = tid; i < n; i += 256) m = fmaxf(m, s[i]);
  red[tid] = m; __syncthreads();
  for (int o = 128; o; o >>= 1) { if (tid < o) red[tid] = fmaxf(red[tid], red[tid + o]); __syncthreads(); }
  if (tid == 0) out[0] = red[0];
}

// e = exp(s - mx); den[b] = sum over contiguous batch block of size P
__global__ __launch_bounds__(128) void softmax_den_kernel(float* __restrict__ s, const float* __restrict__ mx,
                                                          float* __restrict__ den, int P) {
  int b = blockIdx.x, tid = threadIdx.x;
  __shared__ float red[128];
  float acc = 0.f;
  for (int p = tid; p < P; p += 128) {
    float e = __expf(s[(size_t)b * P + p] - mx[0]);
    s[(size_t)b * P + p] = e; acc += e;
  }
  red[tid] = acc; __syncthreads();
  for (int o = 64; o; o >>= 1) { if (tid < o) red[tid] += red[tid + o]; __syncthreads(); }
  if (tid == 0) den[b] = red[0];
}

__global__ __launch_bounds__(256) void pool_weighted_kernel(const float* __restrict__ x, const float* __restrict__ e,
                                                            const float* __restrict__ den, float* __restrict__ out, int P) {
  int b = blockIdx.x, tid = threadIdx.x;
  float inv = 1.f / (den[b] + 1e-8f);
  for (int d = tid; d < DD_; d += 256) {
    float acc = 0.f;
    for (int p = 0; p < P; ++p) acc += e[(size_t)b * P + p] * x[((size_t)b * P + p) * DD_ + d];
    out[(size_t)b * DD_ + d] = acc * inv;
  }
}

__global__ __launch_bounds__(256) void pool_mean_kernel(const float* __restrict__ x, float* __restrict__ out, int P) {
  int b = blockIdx.x, tid = threadIdx.x;
  float inv = 1.f / (float)P;
  for (int d = tid; d < DD_; d += 256) {
    float acc = 0.f;
    for (int p = 0; p < P; ++p) acc += x[((size_t)b * P + p) * DD_ + d];
    out[(size_t)b * DD_ + d] = acc * inv;
  }
}

// fusion cross-attention: block per (batch, head); 1 query x 384 keys
__global__ __launch_bounds__(128) void fusion_attn_kernel(
    const float* __restrict__ Q, const float* __restrict__ Kb,
    const float* __restrict__ Vb, float* __restrict__ O)
{
  const int b = blockIdx.x >> 2, hh = blockIdx.x & 3;
  const int tid = threadIdx.x;
  __shared__ float q[DH_], sc[MAXN_], red[128];
  q[tid] = Q[(size_t)b * DD_ + hh * DH_ + tid];
  __syncthreads();
  for (int kk = tid; kk < MAXN_; kk += 128) {
    int r = (kk < PFC) ? (b * PFC + kk) : (NF_ + b * POC + (kk - PFC));
    const float* kp = Kb + (size_t)r * DD_ + hh * DH_;
    float acc = 0.f;
    for (int d = 0; d < DH_; ++d) acc += q[d] * kp[d];
    sc[kk] = acc * 0.0883883476483184f;   // 1/sqrt(128)
  }
  __syncthreads();
  float m = -3.4e38f;
  for (int kk = tid; kk < MAXN_; kk += 128) m = fmaxf(m, sc[kk]);
  red[tid] = m; __syncthreads();
  for (int o = 64; o; o >>= 1) { if (tid < o) red[tid] = fmaxf(red[tid], red[tid + o]); __syncthreads(); }
  const float mx = red[0]; __syncthreads();
  float sume = 0.f;
  for (int kk = tid; kk < MAXN_; kk += 128) { float e = __expf(sc[kk] - mx); sc[kk] = e; sume += e; }
  red[tid] = sume; __syncthreads();
  for (int o = 64; o; o >>= 1) { if (tid < o) red[tid] += red[tid + o]; __syncthreads(); }
  const float invd = 1.f / red[0]; __syncthreads();
  float acc = 0.f;
  for (int p = 0; p < MAXN_; ++p) {
    int r = (p < PFC) ? (b * PFC + p) : (NF_ + b * POC + (p - PFC));
    acc += sc[p] * Vb[(size_t)r * DD_ + hh * DH_ + tid];
  }
  O[(size_t)b * DD_ + hh * DH_ + tid] = acc * invd;
}

// out = Hf + alpha[d] * fused[row/PF, d]
__global__ void ec_combine_kernel(const float* __restrict__ Hf, const float* __restrict__ alpha,
                                  const float* __restrict__ fused, float* __restrict__ out, size_t n) {
  size_t i = (size_t)blockIdx.x * 256 + threadIdx.x;
  if (i >= n) return;
  int d = (int)(i % DD_);
  int b = (int)((i / DD_) / PFC);
  out[i] = Hf[i] + alpha[d] * fused[(size_t)b * DD_ + d];
}

// y += lam[0] * x ; optional bf16 mirror
__global__ void axpy_bf_kernel(float* __restrict__ y, const float* __restrict__ x,
                               const float* __restrict__ lam, unsigned short* __restrict__ ybf, size_t n) {
  size_t i = (size_t)blockIdx.x * 256 + threadIdx.x;
  if (i >= n) return;
  float v = y[i] + lam[0] * x[i];
  y[i] = v;
  if (ybf) ybf[i] = f2bf(v);
}

// comb = [a | b | c] -> bf16
__global__ void concat3_bf_kernel(const float* __restrict__ a, const float* __restrict__ b,
                                  const float* __restrict__ c, unsigned short* __restrict__ out) {
  int i = blockIdx.x * 256 + threadIdx.x;
  if (i >= BB_ * 3 * DD_) return;
  int row = i / (3 * DD_), col = i % (3 * DD_);
  float v = (col < DD_) ? a[row * DD_ + col]
          : (col < 2 * DD_) ? b[row * DD_ + col - DD_]
          : c[row * DD_ + col - 2 * DD_];
  out[i] = f2bf(v);
}

// tiny classifier head: out[b,c] = dot(in[b,:Dk], W[c,:Dk]) + bias[c]
__global__ void small_linear_kernel(const float* __restrict__ in, const float* __restrict__ W,
                                    const float* __restrict__ bias, float* __restrict__ out, int Dk) {
  int b = blockIdx.x, tid = threadIdx.x;
  int c = tid >> 5, lane = tid & 31;
  if (c >= NC_) return;
  float acc = 0.f;
  for (int d = lane; d < Dk; d += 32) acc += in[(size_t)b * Dk + d] * W[(size_t)c * Dk + d];
  for (int o = 16; o; o >>= 1) acc += __shfl_xor(acc, o, 32);
  if (lane == 0) out[b * NC_ + c] = acc + bias[c];
}

// =====================================================================
// host driver
// =====================================================================
extern "C" void kernel_launch(void* const* d_in, const int* in_sizes, int n_in,
                              void* d_out, int out_size, void* d_ws, size_t ws_size,
                              hipStream_t stream) {
  (void)in_sizes; (void)n_in; (void)out_size; (void)ws_size;
  char* ws = (char*)d_ws;
  constexpr size_t MB = 1ull << 20;

  // workspace layout
  unsigned short* WB     = (unsigned short*)(ws + 0);        // 8MB weight bf16 scratch
  unsigned short* BFBIG  = (unsigned short*)(ws + 8 * MB);   // 128MB bf16 activations
  unsigned short* HFB    = BFBIG;                            // face h bf16 (16MB)
  unsigned short* HOB    = (unsigned short*)(ws + 24 * MB);  // ctx h bf16 (32MB)
  float* TMP    = (float*)(ws + 136 * MB);                   // 64MB
  float* XL     = (float*)(ws + 200 * MB);                   // 64MB
  float* XR     = (float*)(ws + 264 * MB);                   // 64MB
  float* FACEP  = (float*)(ws + 328 * MB);                   // 32MB
  unsigned short* FACEPB = (unsigned short*)(ws + 360 * MB); // 16MB
  float* OBJP   = (float*)(ws + 376 * MB);                   // 64MB
  unsigned short* OBJPB  = (unsigned short*)(ws + 440 * MB); // 32MB
  float* HF     = (float*)(ws + 472 * MB);                   // 32MB
  float* HO     = (float*)(ws + 504 * MB);                   // 64MB
  float* KB     = (float*)(ws + 568 * MB);                   // 96MB
  float* VB     = (float*)(ws + 664 * MB);                   // 96MB
  float* LOGIT  = (float*)(ws + 760 * MB);                   // 2MB
  char*  SMR    = ws + 762 * MB;                             // small region
  unsigned short* SCBF    = (unsigned short*)(SMR + 0 * MB);
  float* SCG    = (float*)(SMR + 1 * MB);
  float* SCENEP = (float*)(SMR + 2 * MB);
  unsigned short* SCENEPB = (unsigned short*)(SMR + 3 * MB);
  float* QB     = (float*)(SMR + 4 * MB);
  float* OB     = (float*)(SMR + 5 * MB);
  unsigned short* OBB = (unsigned short*)(SMR + 6 * MB);
  float* ATTN   = (float*)(SMR + 7 * MB);
  float* FUSED  = (float*)(SMR + 8 * MB);
  float* POOLA  = (float*)(SMR + 9 * MB);
  float* POOLB  = (float*)(SMR + 10 * MB);
  float* SBUF   = (float*)(SMR + 11 * MB);
  float* DEN    = (float*)(SMR + 12 * MB);
  float* SMAX   = (float*)(SMR + 12 * MB + 4096);
  unsigned short* COMBB = (unsigned short*)(SMR + 13 * MB);
  float* ZT     = (float*)(SMR + 14 * MB);
  float* ZB     = (float*)(SMR + 15 * MB);

  // -------- inputs (setup_inputs dict order) --------
  int pi = 0;
  auto F = [&]() { return (const float*)d_in[pi++]; };
  auto I = [&]() { return (const int*)d_in[pi++]; };
  const float* face_x    = F();
  const float* context_x = F();
  const float* scene_x   = F();
  const int* fei = I();  const int* fsrc = fei;        // row 0 = src
  const int* cei = I();  const int* csrc = cei;
  const int* fb = I(); (void)fb;
  const int* cb = I(); (void)cb;
  // params (85 arrays, dict traversal order)
  const float *rf_ln0_g=F(),*rf_ln0_b=F(),*rf_W1=F(),*rf_b1=F(),*rf_ln1_g=F(),*rf_ln1_b=F(),
              *rf_W2=F(),*rf_b2=F(),*rf_ln2_g=F(),*rf_ln2_b=F();
  const float *ro_ln0_g=F(),*ro_ln0_b=F(),*ro_W=F(),*ro_b=F(),*ro_ln1_g=F(),*ro_ln1_b=F();
  const float *rs_ln0_g=F(),*rs_ln0_b=F(),*rs_W=F(),*rs_b=F(),*rs_ln1_g=F(),*rs_ln1_b=F();
  struct GatP { const float *in_W,*in_b,*in_g,*in_bt,*Wl[2],*Wr[2],*att[2],*lg[2],*lb[2]; };
  GatP fg, cg;
  fg.in_W=F(); fg.in_b=F(); fg.in_g=F(); fg.in_bt=F();
  for (int l=0;l<2;++l){ fg.Wl[l]=F(); fg.Wr[l]=F(); fg.att[l]=F(); fg.lg[l]=F(); fg.lb[l]=F(); }
  cg.in_W=F(); cg.in_b=F(); cg.in_g=F(); cg.in_bt=F();
  for (int l=0;l<2;++l){ cg.Wl[l]=F(); cg.Wr[l]=F(); cg.att[l]=F(); cg.lg[l]=F(); cg.lb[l]=F(); }
  const float *pb_W1=F(),*pb_b1=F(),*pb_W2=F(),*pb_b2=F();
  const float *pf_W1=F(),*pf_b1=F(),*pf_W2=F(),*pf_b2=F();
  const float *cfW=F(),*cfb=F(),*ccW=F(),*ccb=F(),*csW=F(),*csb=F();
  const float *fu_in_w=F(),*fu_in_b=F(),*fu_out_w=F(),*fu_out_b=F(),*fu_ln_g=F(),*fu_ln_b=F();
  const float *ec_alpha=F(),*ec_ln_g=F(),*ec_ln_b=F();
  const float *lam_face=F(),*lam_obj=F();
  const float *rawf_W=F(),*rawf_b=F(),*rawo_W=F(),*rawo_b=F();
  const float *cw_W1=F(),*cw_b1=F(),*cw_ln_g=F(),*cw_ln_b=F(),*cw_W2=F(),*cw_b2=F();
  float* out = (float*)d_out;

  // -------- helpers --------
  auto cvt = [&](const float* x, unsigned short* y, size_t n) {
    cvt_bf16_kernel<<<dim3((unsigned)((n + 255) / 256)), 256, 0, stream>>>(x, y, n);
  };
  auto ln = [&](const float* x, const float* xadd, const float* g, const float* b,
                const float* res, float* yf, unsigned short* ybf, int rows, int cols, int act) {
    ln_kernel<<<dim3(rows), 256, 0, stream>>>(x, xadd, g, b, res, yf, ybf, cols, act);
  };
  auto gemm = [&](const unsigned short* A, const float* Wf, const float* bias, float* C,
                  int M, int N, int K, int act) {
    cvt(Wf, WB, (size_t)N * K);
    dim3 g((N + TBN - 1) / TBN, (M + TBM - 1) / TBM);
    gemm_bf16_kernel<<<g, 256, 0, stream>>>(A, WB, bias, C, M, N, K, act);
  };
  auto rungat = [&](const GatP& gp, const unsigned short* xbf, const int* srcarr, int Nn,
                    float* H, unsigned short* Hbf) {
    gemm(xbf, gp.in_W, gp.in_b, TMP, Nn, DD_, DD_, 0);
    ln(TMP, nullptr, gp.in_g, gp.in_bt, nullptr, H, Hbf, Nn, DD_, 1);
    int Ereal = Nn * KK_, Etot = Ereal + Nn;
    for (int l = 0; l < 2; ++l) {
      gemm(Hbf, gp.Wl[l], nullptr, XL, Nn, DD_, DD_, 0);
      gemm(Hbf, gp.Wr[l], nullptr, XR, Nn, DD_, DD_, 0);
      size_t thr = (size_t)Etot * HH_ * 32;
      gat_logit_kernel<<<dim3((unsigned)((thr + 255) / 256)), 256, 0, stream>>>(XL, XR, srcarr, gp.att[l], LOGIT, Nn, Ereal);
      gat_agg_kernel<<<dim3(Nn), 128, 0, stream>>>(XL, srcarr, LOGIT, TMP, Nn, Ereal);
      ln(TMP, nullptr, gp.lg[l], gp.lb[l], H, H, Hbf, Nn, DD_, 2);  // h += elu(LN(agg))
    }
  };

  // ================= forward =================
  // rf: face 4096 -> 1024 -> 512
  ln(face_x, nullptr, rf_ln0_g, rf_ln0_b, nullptr, nullptr, BFBIG, NF_, 4096, 0);
  gemm(BFBIG, rf_W1, rf_b1, TMP, NF_, 1024, 4096, 0);
  ln(TMP, nullptr, rf_ln1_g, rf_ln1_b, nullptr, nullptr, BFBIG, NF_, 1024, 1);
  gemm(BFBIG, rf_W2, rf_b2, TMP, NF_, DD_, 1024, 0);
  ln(TMP, nullptr, rf_ln2_g, rf_ln2_b, nullptr, FACEP, FACEPB, NF_, DD_, 1);
  // ro: objects 2048 -> 512
  ln(context_x, nullptr, ro_ln0_g, ro_ln0_b, nullptr, nullptr, BFBIG, NO_, 2048, 0);
  gemm(BFBIG, ro_W, ro_b, TMP, NO_, DD_, 2048, 0);
  ln(TMP, nullptr, ro_ln1_g, ro_ln1_b, nullptr, OBJP, OBJPB, NO_, DD_, 1);
  // rs: scene 1024 -> 512
  ln(scene_x, nullptr, rs_ln0_g, rs_ln0_b, nullptr, nullptr, SCBF, BB_, 1024, 0);
  gemm(SCBF, rs_W, rs_b, SCG, BB_, DD_, 1024, 0);
  ln(SCG, nullptr, rs_ln1_g, rs_ln1_b, nullptr, SCENEP, SCENEPB, BB_, DD_, 1);
  // GAT stacks
  rungat(fg, FACEPB, fsrc, NF_, HF, HFB);
  rungat(cg, OBJPB,  csrc, NO_, HO, HOB);
  // out_face: attn_pool(Hf) -> clf
  gemm(HFB, pb_W1, pb_b1, TMP, NF_, DH_, DD_, 2);                       // tanh
  rowdot_kernel<<<dim3((NF_ + 7) / 8), 256, 0, stream>>>(TMP, pb_W2, pb_b2, SBUF, NF_, DH_);
  max_reduce_kernel<<<dim3(1), 256, 0, stream>>>(SBUF, NF_, SMAX);
  softmax_den_kernel<<<dim3(BB_), 128, 0, stream>>>(SBUF, SMAX, DEN, PFC);
  pool_weighted_kernel<<<dim3(BB_), 256, 0, stream>>>(HF, SBUF, DEN, POOLA, PFC);
  small_linear_kernel<<<dim3(BB_), 96, 0, stream>>>(POOLA, cfW, cfb, out + 0, DD_);
  // out_ctx: mean_pool(Ho) -> clf
  pool_mean_kernel<<<dim3(BB_), 256, 0, stream>>>(HO, POOLB, POC);
  small_linear_kernel<<<dim3(BB_), 96, 0, stream>>>(POOLB, ccW, ccb, out + BB_ * NC_, DD_);
  // out_scene
  small_linear_kernel<<<dim3(BB_), 96, 0, stream>>>(SCENEP, csW, csb, out + 2 * BB_ * NC_, DD_);
  // fusion: Q from scene, K/V from [Hf;Ho]
  gemm(SCENEPB, fu_in_w,               fu_in_b,        QB,             BB_, DD_, DD_, 0);
  gemm(HFB, fu_in_w + (size_t)DD_ * DD_,     fu_in_b + DD_,     KB,             NF_, DD_, DD_, 0);
  gemm(HOB, fu_in_w + (size_t)DD_ * DD_,     fu_in_b + DD_,     KB + (size_t)NF_ * DD_, NO_, DD_, DD_, 0);
  gemm(HFB, fu_in_w + (size_t)2 * DD_ * DD_, fu_in_b + 2 * DD_, VB,             NF_, DD_, DD_, 0);
  gemm(HOB, fu_in_w + (size_t)2 * DD_ * DD_, fu_in_b + 2 * DD_, VB + (size_t)NF_ * DD_, NO_, DD_, DD_, 0);
  fusion_attn_kernel<<<dim3(BB_ * HH_), 128, 0, stream>>>(QB, KB, VB, OB);
  cvt(OB, OBB, (size_t)BB_ * DD_);
  gemm(OBB, fu_out_w, fu_out_b, ATTN, BB_, DD_, DD_, 0);
  ln(ATTN, SCENEP, fu_ln_g, fu_ln_b, nullptr, FUSED, nullptr, BB_, DD_, 0);  // LN(q + attn_out)
  // ec + raw residuals
  ec_combine_kernel<<<dim3((unsigned)(((size_t)NF_ * DD_ + 255) / 256)), 256, 0, stream>>>(HF, ec_alpha, FUSED, XL, (size_t)NF_ * DD_);
  ln(XL, nullptr, ec_ln_g, ec_ln_b, nullptr, HF, nullptr, NF_, DD_, 0);
  gemm(FACEPB, rawf_W, rawf_b, XR, NF_, DD_, DD_, 0);
  axpy_bf_kernel<<<dim3((unsigned)(((size_t)NF_ * DD_ + 255) / 256)), 256, 0, stream>>>(HF, XR, lam_face, HFB, (size_t)NF_ * DD_);
  gemm(OBJPB, rawo_W, rawo_b, XL, NO_, DD_, DD_, 0);
  axpy_bf_kernel<<<dim3((unsigned)(((size_t)NO_ * DD_ + 255) / 256)), 256, 0, stream>>>(HO, XL, lam_obj, nullptr, (size_t)NO_ * DD_);
  // feat_face (pool_final) / feat_obj
  gemm(HFB, pf_W1, pf_b1, TMP, NF_, DH_, DD_, 2);                       // tanh
  rowdot_kernel<<<dim3((NF_ + 7) / 8), 256, 0, stream>>>(TMP, pf_W2, pf_b2, SBUF, NF_, DH_);
  max_reduce_kernel<<<dim3(1), 256, 0, stream>>>(SBUF, NF_, SMAX);
  softmax_den_kernel<<<dim3(BB_), 128, 0, stream>>>(SBUF, SMAX, DEN, PFC);
  pool_weighted_kernel<<<dim3(BB_), 256, 0, stream>>>(HF, SBUF, DEN, POOLA, PFC);
  pool_mean_kernel<<<dim3(BB_), 256, 0, stream>>>(HO, POOLB, POC);
  // out_whole
  concat3_bf_kernel<<<dim3((BB_ * 3 * DD_ + 255) / 256), 256, 0, stream>>>(POOLA, POOLB, FUSED, COMBB);
  gemm(COMBB, cw_W1, cw_b1, ZT, BB_, DD_, 3 * DD_, 0);
  ln(ZT, nullptr, cw_ln_g, cw_ln_b, nullptr, ZB, nullptr, BB_, DD_, 1);
  small_linear_kernel<<<dim3(BB_), 96, 0, stream>>>(ZB, cw_W2, cw_b2, out + 3 * BB_ * NC_, DD_);
}
